// TensorProduct_56083682951306
// MI455X (gfx1250) — matrix operand order repaired and checked by reading, not compiled, not run on hardware
//
#include <hip/hip_runtime.h>

// ---------------------------------------------------------------------------
// Spherical-harmonic tensor product on MI455X (gfx1250).
//
//   out[n, p, l3, f] = sum_{l1m,l2m} x1[n, pa, l1m, f] * x2[n, pb, l2m, f]
//                      * cg[l1m, l2m, l3]         (4 couples -> 2 parities)
//
// Per node:  D[25(+7) x 128] = CG2[25 x 96] * (Bcouple1 + Bcouple2)[96 x 128]
// CG2 held in VGPRs as 6 f16 WMMA A-fragments (3 K-chunks x 2 M-tiles),
// products built on the fly with packed-f16 mul/fma, f32 accumulation via
// v_wmma_f32_16x16x32_f16.
//
// K permutation (our choice; A and B agree): pair slot s = c*8+v in [0,24),
// holding halves h=0/1:
//   s<20 : (l1 = 2*(s&3)+h, l2pair t = s>>2)     -> only s1 needs cndmask
//   s=20 : (l1 = 8, t = h)                        -> only s2 needs cndmask
//   s=21 : (l1 = 8, t = 2+h)                      -> only s2 needs cndmask
//   s=22 : h==0 -> (8,4), h==1 -> zero pad        -> zero-select on s1
//   s=23 : all padding (no math)
// Memory roofline: ~2.2 GB / 23.3 TB/s ~ 95 us; streaming NT hints on x/out.
// ---------------------------------------------------------------------------

typedef __attribute__((ext_vector_type(16))) _Float16 v16h;
typedef __attribute__((ext_vector_type(8)))  float    v8f;
typedef __attribute__((ext_vector_type(2)))  _Float16 h2;

#define L1DIM 9
#define L3DIM 25
#define FDIM  128

static __device__ __forceinline__ h2 pkrtz(float a, float b) {
    // builtin returns __fp16x2; bit-cast to _Float16x2 (same layout)
    auto r = __builtin_amdgcn_cvt_pkrtz(a, b);   // v_cvt_pk_rtz_f16_f32
    union U { decltype(r) s; h2 d; } u;
    u.s = r;
    return u.d;
}

static __device__ __forceinline__ h2 hzero() {
    h2 r; r[0] = (_Float16)0.0f; r[1] = (_Float16)0.0f; return r;
}

// bit-exact select on a packed f16 pair (compiles to v_cndmask_b32)
static __device__ __forceinline__ h2 selh2(bool c, h2 a, h2 b) {
    union U { h2 v; unsigned u; };
    U ua, ub, r; ua.v = a; ub.v = b;
    r.u = c ? ua.u : ub.u;
    return r.v;
}

__global__ __launch_bounds__(256)
void tp_cg_wmma_kernel(const float* __restrict__ x1,
                       const float* __restrict__ x2,
                       const float* __restrict__ cg,
                       float* __restrict__ out,
                       int nNodes)
{
    const int lane  = threadIdx.x & 31;
    const int wave  = threadIdx.x >> 5;          // 8 waves -> 8 f-slices of 16
    const int fcol  = wave * 16 + (lane & 15);   // this lane's channel
    const bool hi   = (lane >= 16);              // lane half selects K range
    const int mrow  = lane & 15;

    // -----------------------------------------------------------------
    // Constant CG A-fragments (once).  A 16x32 fill in this code's K order (not AMD's table order):
    //   lanes 0-15 : K = e;  lanes 16-31: +16 (matches B fill).
    // Decompose K -> (slot s, half hh, low-bit b) per the permutation.
    // -----------------------------------------------------------------
    v16h Afrag[3][2];
    #pragma unroll
    for (int c = 0; c < 3; ++c) {
        #pragma unroll
        for (int t = 0; t < 2; ++t) {
            v16h a;
            #pragma unroll
            for (int e = 0; e < 16; ++e) {
                int kb  = c * 32 + e;
                int k   = hi ? (kb + 16) : kb;         // runtime (lane half)
                int rem = k & 31;
                int hh  = (rem >> 4) & 1;
                int v   = (rem & 15) >> 1;
                int b   = k & 1;
                int s   = c * 8 + v;
                int l1 = 0, l2 = 0;
                bool valid = false;
                if (s < 20)       { l1 = 2*(s & 3) + hh; l2 = 2*(s >> 2) + b; valid = true; }
                else if (s == 20) { l1 = 8; l2 = 2*hh + b;     valid = true; }
                else if (s == 21) { l1 = 8; l2 = 4 + 2*hh + b; valid = true; }
                else if (s == 22) { l1 = 8; l2 = 8; valid = (hh == 0) && (b == 0); }
                int l3 = t * 16 + mrow;
                float vv = 0.0f;
                if (valid && l3 < L3DIM)
                    vv = cg[(l1 * L1DIM + l2) * L3DIM + l3];
                a[e] = (_Float16)vv;
            }
            Afrag[c][t] = a;
        }
    }

    const size_t nodeStride = (size_t)2 * L1DIM * FDIM;   // floats per node

    for (int n = blockIdx.x; n < nNodes; n += gridDim.x) {
        // prefetch next node's inputs (global_prefetch_b8)
        int n2 = n + gridDim.x;
        if (n2 < nNodes) {
            __builtin_prefetch(x1 + (size_t)n2 * nodeStride + fcol, 0, 1);
            __builtin_prefetch(x2 + (size_t)n2 * nodeStride + fcol, 0, 1);
        }

        // ---- load this lane's 36 input values (streaming, NT) -------
        const size_t base = (size_t)n * nodeStride + fcol;
        float ae[L1DIM], ao[L1DIM], be[L1DIM], bo[L1DIM];
        #pragma unroll
        for (int l = 0; l < L1DIM; ++l) {
            ae[l] = __builtin_nontemporal_load(x1 + base + (size_t)l * FDIM);
            ao[l] = __builtin_nontemporal_load(x1 + base + (size_t)(L1DIM + l) * FDIM);
            be[l] = __builtin_nontemporal_load(x2 + base + (size_t)l * FDIM);
            bo[l] = __builtin_nontemporal_load(x2 + base + (size_t)(L1DIM + l) * FDIM);
        }

        // ---- pack to f16 with v_cvt_pk_rtz_f16_f32 ------------------
        h2 b1e[L1DIM], b1o[L1DIM];        // x1 broadcast pairs
        #pragma unroll
        for (int l = 0; l < L1DIM; ++l) {
            b1e[l] = pkrtz(ae[l], ae[l]);
            b1o[l] = pkrtz(ao[l], ao[l]);
        }
        h2 p2e[5], p2o[5];                // x2 adjacent pairs (t = l2 pair)
        #pragma unroll
        for (int t = 0; t < 4; ++t) {
            p2e[t] = pkrtz(be[2 * t], be[2 * t + 1]);
            p2o[t] = pkrtz(bo[2 * t], bo[2 * t + 1]);
        }
        p2e[4] = pkrtz(be[8], 0.0f);
        p2o[4] = pkrtz(bo[8], 0.0f);

        // ---- accumulators: 2 parities x 2 M-tiles -------------------
        v8f acc[2][2];
        #pragma unroll
        for (int p = 0; p < 2; ++p)
            #pragma unroll
            for (int t = 0; t < 2; ++t)
                #pragma unroll
                for (int r = 0; r < 8; ++r)
                    acc[p][t][r] = 0.0f;

        // ---- K loop: 3 chunks of 32; couples fused pre-WMMA ---------
        #pragma unroll
        for (int c = 0; c < 3; ++c) {
            v16h Bev, Bod;
            #pragma unroll
            for (int v = 0; v < 8; ++v) {
                const int s = c * 8 + v;     // compile-time after unroll
                h2 pe, po;
                if (s < 20) {                       // shared l2-pair t
                    const int a = s & 3, t = s >> 2;
                    h2 s1e = selh2(hi, b1e[2*a + 1], b1e[2*a]);
                    h2 s1o = selh2(hi, b1o[2*a + 1], b1o[2*a]);
                    pe = s1e * p2e[t] + s1o * p2o[t];   // ee + oo
                    po = s1e * p2o[t] + s1o * p2e[t];   // eo + oe
                } else if (s == 20 || s == 21) {    // shared l1 = 8
                    const int t0 = (s == 20) ? 0 : 2;
                    h2 s2e = selh2(hi, p2e[t0 + 1], p2e[t0]);
                    h2 s2o = selh2(hi, p2o[t0 + 1], p2o[t0]);
                    pe = b1e[8] * s2e + b1o[8] * s2o;
                    po = b1e[8] * s2o + b1o[8] * s2e;
                } else if (s == 22) {               // (8,4) on lo half only
                    const h2 z = hzero();
                    h2 s1e = selh2(hi, z, b1e[8]);
                    h2 s1o = selh2(hi, z, b1o[8]);
                    pe = s1e * p2e[4] + s1o * p2o[4];
                    po = s1e * p2o[4] + s1o * p2e[4];
                } else {                            // s == 23: all padding
                    pe = hzero(); po = hzero();
                }
                Bev[2*v] = pe[0]; Bev[2*v + 1] = pe[1];
                Bod[2*v] = po[0]; Bod[2*v + 1] = po[1];
            }
            #pragma unroll
            for (int t = 0; t < 2; ++t) {
                acc[0][t] = __builtin_amdgcn_wmma_f32_16x16x32_f16(
                    false, Afrag[c][t], false, Bev, (short)0, acc[0][t],
                    false, false);
                acc[1][t] = __builtin_amdgcn_wmma_f32_16x16x32_f16(
                    false, Afrag[c][t], false, Bod, (short)0, acc[1][t],
                    false, false);
            }
        }

        // ---- store (NT): C/D layout: VGPR r -> row r (+8 hi lanes) --
        const int row0 = hi ? 8 : 0;
        #pragma unroll
        for (int p = 0; p < 2; ++p) {
            const size_t obase =
                ((size_t)n * 2 + p) * (size_t)L3DIM * FDIM + fcol;
            #pragma unroll
            for (int r = 0; r < 8; ++r) {          // tile 0: l3 = 0..15
                int l3 = row0 + r;
                __builtin_nontemporal_store(acc[p][0][r],
                                            out + obase + (size_t)l3 * FDIM);
            }
            #pragma unroll
            for (int r = 0; r < 8; ++r) {          // tile 1: l3 = 16..24
                int l3 = 16 + row0 + r;
                if (l3 < L3DIM)
                    __builtin_nontemporal_store(acc[p][1][r],
                                                out + obase + (size_t)l3 * FDIM);
            }
        }
    }
}

extern "C" void kernel_launch(void* const* d_in, const int* in_sizes, int n_in,
                              void* d_out, int out_size, void* d_ws, size_t ws_size,
                              hipStream_t stream) {
    const float* x1 = (const float*)d_in[0];
    const float* x2 = (const float*)d_in[1];
    const float* cg = (const float*)d_in[2];
    float* out = (float*)d_out;

    const int nNodes = in_sizes[0] / (2 * L1DIM * FDIM);   // 50000
    int blocks = 4096;                                     // grid-stride
    if (blocks > nNodes) blocks = nNodes;
    if (blocks < 1) blocks = 1;

    tp_cg_wmma_kernel<<<blocks, 256, 0, stream>>>(x1, x2, cg, out, nNodes);
}
